// CriticTorsionNet_74543452389455
// MI455X (gfx1250) — compile-verified, matrix operand order checked by reading, not run on hardware
//
#include <hip/hip_runtime.h>
#include <math.h>

typedef _Float16 half_t;
typedef __attribute__((ext_vector_type(16))) _Float16 v16h;
typedef __attribute__((ext_vector_type(8)))  float    v8f;

#define NN 50000     // nodes
#define NE 100000    // edges
#define NB 500       // graphs
#define ETILES (NE/16)   // 6250
#define NTILES (NN/16)   // 3125

#define WMMA(A,B,C) __builtin_amdgcn_wmma_f32_16x16x32_f16(false,(A),false,(B),(short)0,(C),false,false)

// map (vgpr-pair t, lane group) -> K index within a 32-wide chunk (CDNA5 16-bit A/B layout)
__device__ __forceinline__ int kk_of(int t, int group) {
  int v = t >> 1;
  return ((v & 4) ? 16 : 0) + (group << 3) + ((v & 3) << 1) + (t & 1);
}
__device__ __forceinline__ float sigf(float x) { return 1.0f / (1.0f + expf(-x)); }

// ---------------------------------------------------------------- prep kernels

__global__ void k_deg(const int* __restrict__ dst, int* __restrict__ deg) {
  int e = blockIdx.x * 256 + threadIdx.x;
  if (e < NE) atomicAdd(&deg[dst[e]], 1);
}

__global__ void k_gidx(const int* __restrict__ batch, int* __restrict__ gstart,
                       int* __restrict__ gcnt) {
  int n = blockIdx.x * 256 + threadIdx.x;
  if (n >= NN) return;
  int b = batch[n];
  atomicAdd(&gcnt[b], 1);
  atomicMin(&gstart[b], n);
}

__global__ void k_lin0(const float* __restrict__ x, const float* __restrict__ w,
                       const float* __restrict__ bias, float* __restrict__ out) {
  int idx = blockIdx.x * 256 + threadIdx.x;
  if (idx >= NN * 64) return;
  int o = idx & 63, n = idx >> 6;
  float a = bias[o] + x[n*3]*w[o*3] + x[n*3+1]*w[o*3+1] + x[n*3+2]*w[o*3+2];
  out[idx] = fmaxf(a, 0.0f);
}

// h1 = relu(edge_attr @ w1.T + b1), stored fp16 pre-swizzled into WMMA A-layout
__global__ void k_h1(const float* __restrict__ ea, const float* __restrict__ w1,
                     const float* __restrict__ b1, half_t* __restrict__ h1p) {
  int idx = blockIdx.x * 256 + threadIdx.x;
  if (idx >= NE * 128) return;
  int k = idx & 127, e = idx >> 7;
  float a = b1[k];
  #pragma unroll
  for (int j = 0; j < 7; ++j) a += ea[e*7+j] * w1[k*7+j];
  a = fmaxf(a, 0.0f);
  int tile = e >> 4, M = e & 15, kc = k >> 5, kk = k & 31;
  int group = (kk >> 3) & 1;
  int v = ((kk >> 4) & 1) * 4 + ((kk >> 1) & 3);
  int t = v * 2 + (kk & 1);
  int lane = group * 16 + M;
  h1p[(((tile*4 + kc)*32) + lane)*16 + t] = (half_t)a;
}

// mlp_w2 [4096,128] -> fp16 B-layout tiles: [i(64)][kc(4)][ot(4)][lane(32)][16]
__global__ void k_pack_w2(const float* __restrict__ w2, half_t* __restrict__ d) {
  int idx = blockIdx.x * 256 + threadIdx.x;          // 524288 total
  int t = idx & 15, lane = (idx >> 4) & 31, tile = idx >> 9;
  int ot = tile & 3, kc = (tile >> 2) & 3, i = tile >> 4;
  int ncol = lane & 15, group = lane >> 4;
  int k = kc * 32 + kk_of(t, group);
  int o = ot * 16 + ncol;
  d[idx] = (half_t)w2[(i*64 + o)*128 + k];
}

// mlp_b2 [4096] viewed as [i(64), o(64)] -> B-layout tiles: [kcb(2)][ot(4)][lane][16]
__global__ void k_pack_b2(const float* __restrict__ b2, half_t* __restrict__ d) {
  int idx = blockIdx.x * 256 + threadIdx.x;          // 4096 total
  int t = idx & 15, lane = (idx >> 4) & 31, tile = idx >> 9;
  int ot = tile & 3, kcb = tile >> 2;
  int ncol = lane & 15, group = lane >> 4;
  int i = kcb * 32 + kk_of(t, group);
  int o = ot * 16 + ncol;
  d[idx] = (half_t)b2[i*64 + o];
}

// pack a [*,64] weight into fp16 B tiles [kc(2)][ot(ntiles_o)][lane][16]
// transpose=0: B[k,n]=W[k*64+n]   transpose=1: B[k,n]=W[n*64+k]
__global__ void k_pack_b64(const float* __restrict__ W, half_t* __restrict__ d,
                           int ntiles_o, int transpose) {
  int idx = blockIdx.x * 256 + threadIdx.x;
  if (idx >= 2 * ntiles_o * 512) return;
  int t = idx & 15, lane = (idx >> 4) & 31, tile = idx >> 9;
  int ot = tile % ntiles_o, kc = tile / ntiles_o;
  int ncol = lane & 15, group = lane >> 4;
  int k = kc * 32 + kk_of(t, group);
  int n = ot * 16 + ncol;
  d[idx] = (half_t)(transpose ? W[n*64 + k] : W[k*64 + n]);
}

// ------------------------------------------- fused edge-message kernel (WMMA)
// per 16-edge tile: msg[16,64] = (x[src] (x) h1)[16,8192] @ W2'[8192,64] + x@b2'
__global__ __launch_bounds__(256) void k_msg(
    const float* __restrict__ out, const half_t* __restrict__ h1p,
    const half_t* __restrict__ w2b, const half_t* __restrict__ b2b,
    const int* __restrict__ srci, const int* __restrict__ dsti,
    float* __restrict__ agg) {
  __shared__ float xsh[8][16][68];
  __shared__ __attribute__((aligned(32))) half_t bsh[16 * 512];  // 16 KB stage
  int lane = threadIdx.x & 31, w = threadIdx.x >> 5;
  int tile = blockIdx.x * 8 + w;
  bool valid = tile < ETILES;
  int tb = tile * 16;
  int M = lane & 15, group = lane >> 4;

  if (valid) {
    for (int r = 0; r < 16; ++r) {                 // gather x = out[src] rows
      int s = srci[tb + r];
      float2 v2 = *(const float2*)(out + s*64 + lane*2);
      xsh[w][r][lane*2]   = v2.x;
      xsh[w][r][lane*2+1] = v2.y;
    }
  }
  v16h H1[4] = {};
  if (valid) {
    #pragma unroll
    for (int kc = 0; kc < 4; ++kc)
      H1[kc] = *(const v16h*)(h1p + ((tile*4 + kc)*32 + lane)*16);
  }
  v8f acc[4] = {};

  for (int i = 0; i < 64; ++i) {                   // K = i(64) x k(128)
    __syncthreads();
    {                                              // stage B[i] (16 KB) to LDS
      const uint4* s = (const uint4*)(w2b + (size_t)i * 8192);
      uint4* d = (uint4*)bsh;
      #pragma unroll
      for (int j = 0; j < 4; ++j) d[threadIdx.x + j*256] = s[threadIdx.x + j*256];
      if (i < 63) __builtin_prefetch(w2b + (size_t)(i+1) * 8192, 0, 0);
    }
    __syncthreads();
    if (valid) {
      half_t xh = (half_t)xsh[w][M][i];
      #pragma unroll
      for (int kc = 0; kc < 4; ++kc) {
        v16h A;
        #pragma unroll
        for (int t = 0; t < 16; ++t) A[t] = H1[kc][t] * xh;   // u = x (x) h1
        #pragma unroll
        for (int ot = 0; ot < 4; ++ot) {
          v16h B = *(const v16h*)(bsh + ((kc*4 + ot)*32 + lane)*16);
          acc[ot] = WMMA(A, B, acc[ot]);
        }
      }
    }
  }
  // bias part:  + x[16,64] @ b2[64,64]
  __syncthreads();
  {
    const uint4* s = (const uint4*)b2b;
    uint4* d = (uint4*)bsh;
    #pragma unroll
    for (int j = 0; j < 2; ++j) d[threadIdx.x + j*256] = s[threadIdx.x + j*256];
  }
  __syncthreads();
  if (valid) {
    #pragma unroll
    for (int kcb = 0; kcb < 2; ++kcb) {
      v16h A;
      #pragma unroll
      for (int t = 0; t < 16; ++t)
        A[t] = (half_t)xsh[w][M][kcb*32 + kk_of(t, group)];
      #pragma unroll
      for (int ot = 0; ot < 4; ++ot) {
        v16h B = *(const v16h*)(bsh + ((kcb*4 + ot)*32 + lane)*16);
        acc[ot] = WMMA(A, B, acc[ot]);
      }
    }
    int ncol = lane & 15, mofs = group * 8;
    #pragma unroll
    for (int r = 0; r < 8; ++r) {                  // scatter msg -> agg[dst]
      int n = dsti[tb + mofs + r];
      #pragma unroll
      for (int ot = 0; ot < 4; ++ot)
        atomicAdd(&agg[n*64 + ot*16 + ncol], acc[ot][r]);
    }
  }
}

// ------------------------- node update: conv_root + GRU, all GEMMs via WMMA
__global__ __launch_bounds__(256) void k_node(
    const float* __restrict__ agg, const int* __restrict__ deg,
    float* __restrict__ out,
    const half_t* __restrict__ crb, const half_t* __restrict__ wihb,
    const half_t* __restrict__ whhb,
    const float* __restrict__ conv_bias, const float* __restrict__ b_ih,
    const float* __restrict__ b_hh) {
  __shared__ float ssh[8][16 * 68];
  int lane = threadIdx.x & 31, w = threadIdx.x >> 5;
  int tile = blockIdx.x * 8 + w;
  if (tile >= NTILES) return;                      // no barriers below: safe
  int tb = tile * 16;
  int M = lane & 15, group = lane >> 4, ncol = M, mofs = group * 8;
  float* S = ssh[w];

  for (int r = 0; r < 16; ++r) {                   // stage out (=h) rows
    float2 v2 = *(const float2*)(out + (tb + r)*64 + lane*2);
    S[r*68 + lane*2]   = v2.x;
    S[r*68 + lane*2+1] = v2.y;
  }
  v16h Ah[2];                                      // A-layout of h/out tile
  #pragma unroll
  for (int kc = 0; kc < 2; ++kc)
    #pragma unroll
    for (int t = 0; t < 16; ++t)
      Ah[kc][t] = (half_t)S[M*68 + kc*32 + kk_of(t, group)];
  float hold[4][8];                                // h_old in C layout
  #pragma unroll
  for (int ot = 0; ot < 4; ++ot)
    #pragma unroll
    for (int r = 0; r < 8; ++r)
      hold[ot][r] = S[(mofs + r)*68 + ot*16 + ncol];

  v8f accm[4] = {};                                // m = out @ conv_root
  #pragma unroll
  for (int kc = 0; kc < 2; ++kc)
    #pragma unroll
    for (int ot = 0; ot < 4; ++ot) {
      v16h B = *(const v16h*)(crb + ((kc*4 + ot)*32 + lane)*16);
      accm[ot] = WMMA(Ah[kc], B, accm[ot]);
    }
  #pragma unroll
  for (int r = 0; r < 8; ++r) {                    // m = relu(conv + agg/deg + b)
    int node = tb + mofs + r;
    float invd = 1.0f / fmaxf((float)deg[node], 1.0f);
    #pragma unroll
    for (int ot = 0; ot < 4; ++ot) {
      int o = ot*16 + ncol;
      float mv = accm[ot][r] + agg[node*64 + o] * invd + conv_bias[o];
      S[(mofs + r)*68 + o] = fmaxf(mv, 0.0f);
    }
  }
  v16h Am[2];                                      // A-layout of m tile
  #pragma unroll
  for (int kc = 0; kc < 2; ++kc)
    #pragma unroll
    for (int t = 0; t < 16; ++t)
      Am[kc][t] = (half_t)S[M*68 + kc*32 + kk_of(t, group)];

  float rg[4][8], zg[4][8];
  #pragma unroll
  for (int g = 0; g < 3; ++g) {                    // GRU gates r, z, n
    v8f gi[4] = {}, gh[4] = {};
    #pragma unroll
    for (int ot = 0; ot < 4; ++ot)
      #pragma unroll
      for (int kc = 0; kc < 2; ++kc) {
        v16h Bi = *(const v16h*)(wihb + ((kc*12 + g*4 + ot)*32 + lane)*16);
        gi[ot] = WMMA(Am[kc], Bi, gi[ot]);
        v16h Bh = *(const v16h*)(whhb + ((kc*12 + g*4 + ot)*32 + lane)*16);
        gh[ot] = WMMA(Ah[kc], Bh, gh[ot]);
      }
    #pragma unroll
    for (int ot = 0; ot < 4; ++ot)
      #pragma unroll
      for (int r = 0; r < 8; ++r) {
        int o = g*64 + ot*16 + ncol;
        float xi = gi[ot][r] + b_ih[o];
        float xh = gh[ot][r] + b_hh[o];
        if (g == 0) rg[ot][r] = sigf(xi + xh);
        else if (g == 1) zg[ot][r] = sigf(xi + xh);
        else {
          float nn = tanhf(xi + rg[ot][r] * xh);
          float hn = (1.0f - zg[ot][r]) * nn + zg[ot][r] * hold[ot][r];
          out[(tb + mofs + r)*64 + ot*16 + ncol] = hn;
        }
      }
  }
}

// --------------- Set2Set: per-graph block, all 6 steps, deterministic
__global__ __launch_bounds__(256) void k_s2s(
    const float* __restrict__ out, const int* __restrict__ gstart,
    const int* __restrict__ gcnt,
    const float* __restrict__ w_ih, const float* __restrict__ w_hh,
    const float* __restrict__ b_ih, const float* __restrict__ b_hh,
    float* __restrict__ ebuf, float* __restrict__ qstar) {
  int b = blockIdx.x, j = threadIdx.x;
  __shared__ float qst[128], hsv[64], csv[64], g[256], red[256], rvec[64];
  if (j < 128) qst[j] = 0.0f;
  if (j < 64) { hsv[j] = 0.0f; csv[j] = 0.0f; }
  int cnt = gcnt[b];
  int n0 = (cnt > 0) ? gstart[b] : 0;
  for (int step = 0; step < 6; ++step) {
    __syncthreads();
    float a = b_ih[j] + b_hh[j];                   // LSTM gates [256]
    for (int k = 0; k < 128; ++k) a += qst[k] * w_ih[j*128 + k];
    for (int k = 0; k < 64;  ++k) a += hsv[k] * w_hh[j*64 + k];
    g[j] = a;
    __syncthreads();
    if (j < 64) {
      float c = sigf(g[64+j]) * csv[j] + sigf(g[j]) * tanhf(g[128+j]);
      csv[j] = c;
      hsv[j] = sigf(g[192+j]) * tanhf(c);
    }
    __syncthreads();
    float mx = -INFINITY;                          // e = out . q, block max
    for (int t = j; t < cnt; t += 256) {
      int n = n0 + t;
      float e = 0.0f;
      for (int k = 0; k < 64; ++k) e += out[n*64 + k] * hsv[k];
      ebuf[n] = e;
      mx = fmaxf(mx, e);
    }
    red[j] = mx; __syncthreads();
    for (int s = 128; s > 0; s >>= 1) {
      if (j < s) red[j] = fmaxf(red[j], red[j+s]);
      __syncthreads();
    }
    float emx = red[0]; __syncthreads();
    float sm = 0.0f;                               // softmax denom
    for (int t = j; t < cnt; t += 256) {
      int n = n0 + t;
      float aa = expf(ebuf[n] - emx);
      ebuf[n] = aa;
      sm += aa;
    }
    red[j] = sm; __syncthreads();
    for (int s = 128; s > 0; s >>= 1) {
      if (j < s) red[j] += red[j+s];
      __syncthreads();
    }
    float asum = red[0]; __syncthreads();
    if (j < 64) {                                  // r = sum a*out / asum
      float r = 0.0f;
      for (int t = 0; t < cnt; ++t) r += ebuf[n0 + t] * out[(n0 + t)*64 + j];
      rvec[j] = (cnt > 0) ? r / asum : 0.0f;
    }
    __syncthreads();
    if (j < 128) qst[j] = (j < 64) ? hsv[j] : rvec[j - 64];
  }
  __syncthreads();
  if (j < 128) qstar[b*128 + j] = qst[j];
}

// --------------- memory LSTM (h0=0) + lin1 + lin3 head
__global__ __launch_bounds__(256) void k_final(
    const float* __restrict__ qstar, const float* __restrict__ w_ih,
    const float* __restrict__ b_ih, const float* __restrict__ b_hh,
    const float* __restrict__ lin1_w, const float* __restrict__ lin1_b,
    const float* __restrict__ lin3_w, const float* __restrict__ lin3_b,
    float* __restrict__ dout) {
  int b = blockIdx.x, j = threadIdx.x;
  __shared__ float qs[128], g[512], hm[128], o1[64];
  if (j < 128) qs[j] = qstar[b*128 + j];
  __syncthreads();
  for (int jj = j; jj < 512; jj += 256) {
    float a = b_ih[jj] + b_hh[jj];
    for (int k = 0; k < 128; ++k) a += qs[k] * w_ih[jj*128 + k];
    g[jj] = a;
  }
  __syncthreads();
  if (j < 128) {
    float c = sigf(g[j]) * tanhf(g[256 + j]);      // ii, gg (c0 = 0)
    float h = sigf(g[384 + j]) * tanhf(c);         // oo
    hm[j] = h;
    dout[NB + b*128 + j] = h;                      // hmem
    dout[NB + NB*128 + b*128 + j] = c;             // c
  }
  __syncthreads();
  if (j < 64) {
    float a = lin1_b[j];
    for (int k = 0; k < 128; ++k) a += hm[k] * lin1_w[j*128 + k];
    o1[j] = fmaxf(a, 0.0f);
  }
  __syncthreads();
  if (j == 0) {
    float a = lin3_b[0];
    for (int k = 0; k < 64; ++k) a += o1[k] * lin3_w[k];
    dout[b] = a;                                   // v
  }
}

// ---------------------------------------------------------------- launch
extern "C" void kernel_launch(void* const* d_in, const int* in_sizes, int n_in,
                              void* d_out, int out_size, void* d_ws, size_t ws_size,
                              hipStream_t stream) {
  (void)in_sizes; (void)n_in; (void)out_size; (void)ws_size;
  const float* x        = (const float*)d_in[0];
  const int*   ei       = (const int*)d_in[1];
  const float* ea       = (const float*)d_in[2];
  const int*   batch    = (const int*)d_in[3];
  const float* lin0_w   = (const float*)d_in[4];
  const float* lin0_b   = (const float*)d_in[5];
  const float* mlp_w1   = (const float*)d_in[6];
  const float* mlp_b1   = (const float*)d_in[7];
  const float* mlp_w2   = (const float*)d_in[8];
  const float* mlp_b2   = (const float*)d_in[9];
  const float* conv_root= (const float*)d_in[10];
  const float* conv_bias= (const float*)d_in[11];
  const float* gru_w_ih = (const float*)d_in[12];
  const float* gru_w_hh = (const float*)d_in[13];
  const float* gru_b_ih = (const float*)d_in[14];
  const float* gru_b_hh = (const float*)d_in[15];
  const float* s2s_w_ih = (const float*)d_in[16];
  const float* s2s_w_hh = (const float*)d_in[17];
  const float* s2s_b_ih = (const float*)d_in[18];
  const float* s2s_b_hh = (const float*)d_in[19];
  const float* mem_w_ih = (const float*)d_in[20];
  const float* mem_b_ih = (const float*)d_in[22];
  const float* mem_b_hh = (const float*)d_in[23];
  const float* lin1_w   = (const float*)d_in[24];
  const float* lin1_b   = (const float*)d_in[25];
  const float* lin3_w   = (const float*)d_in[26];
  const float* lin3_b   = (const float*)d_in[27];
  const int* srci = ei;
  const int* dsti = ei + NE;

  char* ws = (char*)d_ws;
  size_t off = 0;
  auto alloc = [&](size_t bytes) -> void* {
    void* p = ws + off;
    off = (off + bytes + 255) & ~(size_t)255;
    return p;
  };
  float*  outb  = (float*)alloc((size_t)NN * 64 * 4);
  float*  agg   = (float*)alloc((size_t)NN * 64 * 4);
  int*    deg   = (int*)alloc((size_t)NN * 4);
  half_t* h1p   = (half_t*)alloc((size_t)NE * 128 * 2);
  half_t* w2b   = (half_t*)alloc((size_t)524288 * 2);
  half_t* b2b   = (half_t*)alloc((size_t)4096 * 2);
  half_t* crb   = (half_t*)alloc((size_t)4096 * 2);
  half_t* wihb  = (half_t*)alloc((size_t)12288 * 2);
  half_t* whhb  = (half_t*)alloc((size_t)12288 * 2);
  float*  ebuf  = (float*)alloc((size_t)NN * 4);
  float*  qstar = (float*)alloc((size_t)NB * 128 * 4);
  int*    gstart= (int*)alloc((size_t)NB * 4);
  int*    gcnt  = (int*)alloc((size_t)NB * 4);

  hipMemsetAsync(deg, 0, (size_t)NN * 4, stream);
  hipMemsetAsync(gcnt, 0, (size_t)NB * 4, stream);
  hipMemsetAsync(gstart, 0x7F, (size_t)NB * 4, stream);

  k_deg <<<(NE + 255)/256, 256, 0, stream>>>(dsti, deg);
  k_gidx<<<(NN + 255)/256, 256, 0, stream>>>(batch, gstart, gcnt);
  k_lin0<<<(NN*64)/256, 256, 0, stream>>>(x, lin0_w, lin0_b, outb);
  k_h1  <<<(NE*128)/256, 256, 0, stream>>>(ea, mlp_w1, mlp_b1, h1p);
  k_pack_w2 <<<2048, 256, 0, stream>>>(mlp_w2, w2b);
  k_pack_b2 <<<16,   256, 0, stream>>>(mlp_b2, b2b);
  k_pack_b64<<<16,   256, 0, stream>>>(conv_root, crb, 4, 0);
  k_pack_b64<<<48,   256, 0, stream>>>(gru_w_ih, wihb, 12, 1);
  k_pack_b64<<<48,   256, 0, stream>>>(gru_w_hh, whhb, 12, 1);

  for (int step = 0; step < 6; ++step) {
    hipMemsetAsync(agg, 0, (size_t)NN * 64 * 4, stream);
    k_msg <<<(ETILES + 7)/8, 256, 0, stream>>>(outb, h1p, w2b, b2b, srci, dsti, agg);
    k_node<<<(NTILES + 7)/8, 256, 0, stream>>>(agg, deg, outb, crb, wihb, whhb,
                                               conv_bias, gru_b_ih, gru_b_hh);
  }

  k_s2s  <<<NB, 256, 0, stream>>>(outb, gstart, gcnt, s2s_w_ih, s2s_w_hh,
                                  s2s_b_ih, s2s_b_hh, ebuf, qstar);
  k_final<<<NB, 256, 0, stream>>>(qstar, mem_w_ih, mem_b_ih, mem_b_hh,
                                  lin1_w, lin1_b, lin3_w, lin3_b, (float*)d_out);
}
